// IntraDaySnapshotEncoder_65000035058009
// MI455X (gfx1250) — matrix operand order repaired and checked by reading, not compiled, not run on hardware
//
#include <hip/hip_runtime.h>
#include <hip/hip_bf16.h>
#include <math.h>

// ============================================================================
// MI455X (gfx1250) implementation of the heterogeneous graph-attention net.
//
// Key algebraic optimization (vs reference): all per-edge [E,128]@[128,128]
// GEMMs are hoisted to per-node GEMMs (gather-after-project), the attention
// score decomposes into per-node scalars (+ per-edge 8-dim term for SS), and
// the SS edge message uses the precomposed 8x128 matrix We@Wm. The remaining
// per-edge work is pure gather/scatter (atomic f32 segment softmax + weighted
// segment sum) -> the kernel set is HBM/L2-bandwidth bound (~2 GB traffic
// ~0.1 ms at 23.3 TB/s), and the dense projections run on v_wmma_f32_16x16x32_f16
// (f16 operands / f32 accumulate; operands are LayerNorm-scale so f16 is safe).
//
// GEMM kernel is templated on <K, SPLIT> so all bounds/split checks constant-
// fold: the A fragment is 4 unconditional global_load_b128 per wave per k-step
// and the k-loop fully unrolls (no predicated scalar loads, no per-element
// s_wait_loadcnt).
//
// Input flattening assumption: params pytree leaves in JAX sorted-dict-key
// order ('W' < 'b'); edge index arrays delivered as int32. 177 flat inputs:
//   0 stock_feat 1 bank_feat 2 industry_feat 3 ea_ss
//   4..11  bank/ind encoders, 12..159 layers (BS,II,IS,SB,SI,SS per layer),
//   160..166 meta, 167..170 stock encoder, 171..176 edge indices.
// Workspace use: ~29 MB of d_ws.  d_out (4000x128 f32) is used as sx storage.
// ============================================================================

#define D_    128
#define NS_   4000
#define NB_   256
#define NI_   128

typedef __attribute__((ext_vector_type(16))) _Float16 v16h;
typedef __attribute__((ext_vector_type(8)))  float    v8f;

// ---------------------------------------------------------------------------
// WMMA GEMM:  Out[M][128] = act( A[M][K] @ W[K][128] + bias )
// SPLIT: concat in K at 128 -> k<128 from A0 (ld 128), k>=128 from A1 (ld 128).
// Block = 256 threads = 8 waves; each wave owns one 16x16 C tile (8 waves
// cover all 128 output columns); grid.x = M/16 (all M here are multiples of 16).
// W is staged transposed into LDS as f16, zero-padded to Kpad (mult of 32).
// K is a template constant: all validity/split conditions constant-fold, the
// k-loop fully unrolls, and A loads are plain 16B vector loads.
// ---------------------------------------------------------------------------
template <int K, bool SPLIT>
__global__ __launch_bounds__(256) void gemm128_wmma(
    const float* __restrict__ A0, const float* __restrict__ A1,
    const float* __restrict__ W, const float* __restrict__ bias,
    float* __restrict__ Out, int act)
{
  constexpr int Kpad = (K + 31) & ~31;
  constexpr int LDA  = SPLIT ? 128 : K;
  __shared__ __align__(64) _Float16 Wt[128 * Kpad];  // <= 64 KB (WGP LDS 320 KB)

  const int tid   = threadIdx.x;
  const int lane  = tid & 31;
  const int wave  = tid >> 5;
  const int half_ = lane >> 4;      // 0: lanes 0-15, 1: lanes 16-31
  const int l16   = lane & 15;

  // Stage W transposed (Wt[n][k]) so each B-fragment is two contiguous 16B ds reads.
  // Global reads are fully coalesced (idx walks W linearly).
  for (int idx = tid; idx < 128 * Kpad; idx += 256) {
    int k = idx >> 7, n = idx & 127;
    Wt[n * Kpad + k] = (k < K) ? (_Float16)W[idx] : (_Float16)0.f;
  }
  __syncthreads();

  const int m0  = blockIdx.x * 16;
  const int row = m0 + l16;                  // A layout: all lanes, M = lane%16
  const int n   = wave * 16 + l16;           // C/B column for this lane
  __builtin_prefetch(A0 + (size_t)row * LDA, 0, 0);   // global_prefetch_b8

  v8f c = {};
#pragma unroll
  for (int k0 = 0; k0 < Kpad; k0 += 32) {
    // A fragment (ISA 7.12.2, 16-bit A 16x32): lanes<16 hold K=k0+[0..7],
    // k0+[16..23]; lanes>=16 hold K=k0+[8..15], k0+[24..31].
    // chunk1 spans [k0, k0+16), chunk2 spans [k0+16, k0+32); K is a multiple
    // of 16 and the SPLIT boundary (128) is chunk-aligned, so each chunk is
    // entirely valid-from-one-matrix or entirely zero (compile-time decision).
    float c1[8], c2[8];
    if (k0 + 16 <= K) {                      // constant-folded
      const float* base = (SPLIT && k0 >= 128)
                            ? A1 + (size_t)row * 128 + (k0 - 128)
                            : A0 + (size_t)row * LDA + k0;
      const float4 u = *(const float4*)(base + half_ * 8);
      const float4 v = *(const float4*)(base + half_ * 8 + 4);
      c1[0]=u.x; c1[1]=u.y; c1[2]=u.z; c1[3]=u.w;
      c1[4]=v.x; c1[5]=v.y; c1[6]=v.z; c1[7]=v.w;
    } else {
#pragma unroll
      for (int t = 0; t < 8; ++t) c1[t] = 0.f;
    }
    if (k0 + 32 <= K) {                      // constant-folded
      const float* base = (SPLIT && k0 + 16 >= 128)
                            ? A1 + (size_t)row * 128 + (k0 + 16 - 128)
                            : A0 + (size_t)row * LDA + (k0 + 16);
      const float4 u = *(const float4*)(base + half_ * 8);
      const float4 v = *(const float4*)(base + half_ * 8 + 4);
      c2[0]=u.x; c2[1]=u.y; c2[2]=u.z; c2[3]=u.w;
      c2[4]=v.x; c2[5]=v.y; c2[6]=v.z; c2[7]=v.w;
    } else {
#pragma unroll
      for (int t = 0; t < 8; ++t) c2[t] = 0.f;
    }
    v16h a;
#pragma unroll
    for (int t = 0; t < 8; ++t) { a[t] = (_Float16)c1[t]; a[8 + t] = (_Float16)c2[t]; }

    // B fragment: lane holds column n, K = k0+[0..7],k0+[16..23] (lanes<16) / k0+[8..15],k0+[24..31].
    typedef __attribute__((ext_vector_type(8))) _Float16 v8h_;
    const v8h_ b0 = *(const v8h_*)&Wt[n * Kpad + k0 + half_ * 8];
    const v8h_ b1 = *(const v8h_*)&Wt[n * Kpad + k0 + 16 + half_ * 8];
    v16h b;
#pragma unroll
    for (int t = 0; t < 8; ++t) { b[t] = b0[t]; b[8 + t] = b1[t]; }
    c = __builtin_amdgcn_wmma_f32_16x16x32_f16(false, a, false, b,
                                               (short)0, c, false, false);
  }

  // C layout: VGPR r, lane l -> row m0 + r + (l>=16 ? 8 : 0), col n.
  const float bv = bias ? bias[n] : 0.f;
#pragma unroll
  for (int r = 0; r < 8; ++r) {
    int rr = m0 + r + half_ * 8;
    float v = c[r] + bv;
    if (act == 1)      v = fmaxf(v, 0.f);
    else if (act == 2) v = tanhf(v);
    Out[(size_t)rr * 128 + n] = v;
  }
}

template <int K, bool SPLIT>
static inline void launch_gemm(const float* A0, const float* A1, const float* W,
                               const float* b, float* Out, int M, int act,
                               hipStream_t st) {
  gemm128_wmma<K, SPLIT><<<dim3(M / 16), dim3(256), 0, st>>>(A0, A1, W, b, Out, act);
}

// ---------------------------------------------------------------------------
__global__ void fill_zero_k(float* p, int n) {
  int i = blockIdx.x * 256 + threadIdx.x;
  if (i < n) p[i] = 0.f;
}

// out[n] = sum_d tanh(X[n][d]) * w[d]   (per-node attention scalar)
__global__ void dot_tanh_k(const float* __restrict__ X, const float* __restrict__ w,
                           float* __restrict__ out, int N) {
  __shared__ float sm[128];
  int nrow = blockIdx.x, d = threadIdx.x;
  sm[d] = tanhf(X[(size_t)nrow * 128 + d]) * w[d];
  __syncthreads();
  for (int s = 64; s > 0; s >>= 1) { if (d < s) sm[d] += sm[d + s]; __syncthreads(); }
  if (d == 0) out[nrow] = sm[0];
}

// Pass A (no edge feats): score_e = a_src[si]+a_dst[di]+b; ex=exp; Z[di]+=ex.
// (scores bounded by |Wa|*tanh range ~6 -> exp without max-shift is safe)
__global__ void edge_scoreA_k(const int* __restrict__ ei, int E,
                              const float* __restrict__ a_src, const float* __restrict__ a_dst,
                              const float* __restrict__ battn,
                              float* __restrict__ exb, float* __restrict__ z) {
  int e = blockIdx.x * 256 + threadIdx.x;
  if (e >= E) return;
  int si = ei[e], di = ei[E + e];
  float ex = __expf(a_src[si] + a_dst[di] + battn[0]);
  exb[e] = ex;
  atomicAdd(&z[di], ex);
}

// Pass A (SS, wave-per-edge): adds per-edge term tanh(ea@We + be) . wa3
__global__ __launch_bounds__(256) void edge_scoreA_ss_k(
    const int* __restrict__ ei, int E, const float* __restrict__ ea,
    const float* __restrict__ We, const float* __restrict__ be,
    const float* __restrict__ wa3, const float* __restrict__ battn,
    const float* __restrict__ a_src, const float* __restrict__ a_dst,
    float* __restrict__ exb, float* __restrict__ z) {
  int e = (blockIdx.x * 256 + threadIdx.x) >> 5;
  int lane = threadIdx.x & 31;
  if (e >= E) return;
  float eal[8];
#pragma unroll
  for (int j = 0; j < 8; ++j) eal[j] = ea[(size_t)e * 8 + j];
  float acc = 0.f;
#pragma unroll
  for (int q = 0; q < 4; ++q) {
    int d = lane * 4 + q;
    float t = be[d];
#pragma unroll
    for (int j = 0; j < 8; ++j) t += eal[j] * We[j * 128 + d];
    acc += tanhf(t) * wa3[d];
  }
#pragma unroll
  for (int off = 16; off > 0; off >>= 1) acc += __shfl_xor(acc, off, 32);
  if (lane == 0) {
    int si = ei[e], di = ei[E + e];
    float ex = __expf(a_src[si] + a_dst[di] + battn[0] + acc);
    exb[e] = ex;
    atomicAdd(&z[di], ex);
  }
}

// Pass B (no edge feats): agg[di] += (ex/Z[di]) * nmsg[si]; thread = (edge, 4 cols)
__global__ void edge_scatterB_k(const int* __restrict__ ei, int E,
                                const float* __restrict__ exb, const float* __restrict__ z,
                                const float* __restrict__ nmsg, float* __restrict__ agg) {
  long gid = (long)blockIdx.x * 256 + threadIdx.x;
  int e = (int)(gid >> 5);
  if (e >= E) return;
  int c0 = ((int)(gid & 31)) * 4;
  int si = ei[e], di = ei[E + e];
  float alpha = exb[e] / z[di];
  const float4 m = *(const float4*)&nmsg[(size_t)si * 128 + c0];
  atomicAdd(&agg[(size_t)di * 128 + c0 + 0], alpha * m.x);
  atomicAdd(&agg[(size_t)di * 128 + c0 + 1], alpha * m.y);
  atomicAdd(&agg[(size_t)di * 128 + c0 + 2], alpha * m.z);
  atomicAdd(&agg[(size_t)di * 128 + c0 + 3], alpha * m.w);
}

// Pass B (SS, wave-per-edge): msg_e = nmsg[si] + ea@Wem + be@Wm
// (wem9 rows 0..7 = We@Wm, row 8 = be@Wm)
__global__ __launch_bounds__(256) void edge_scatterB_ss_k(
    const int* __restrict__ ei, int E, const float* __restrict__ ea,
    const float* __restrict__ wem9, const float* __restrict__ exb,
    const float* __restrict__ z, const float* __restrict__ nmsg,
    float* __restrict__ agg) {
  int e = (blockIdx.x * 256 + threadIdx.x) >> 5;
  int lane = threadIdx.x & 31;
  if (e >= E) return;
  int si = ei[e], di = ei[E + e];
  float alpha = exb[e] / z[di];
  float eal[8];
#pragma unroll
  for (int j = 0; j < 8; ++j) eal[j] = ea[(size_t)e * 8 + j];
#pragma unroll
  for (int q = 0; q < 4; ++q) {
    int c = lane * 4 + q;
    float v = nmsg[(size_t)si * 128 + c] + wem9[8 * 128 + c];
#pragma unroll
    for (int j = 0; j < 8; ++j) v += eal[j] * wem9[j * 128 + c];
    atomicAdd(&agg[(size_t)di * 128 + c], alpha * v);
  }
}

// wem9[j][c] = sum_d We[j][d]*Wm[d][c] (j<8);  wem9[8][c] = sum_d be[d]*Wm[d][c]
__global__ void wem9_k(const float* __restrict__ We, const float* __restrict__ be,
                       const float* __restrict__ Wm, float* __restrict__ wem9) {
  int idx = blockIdx.x * 256 + threadIdx.x;
  if (idx >= 9 * 128) return;
  int j = idx / 128, c = idx - j * 128;
  float s = 0.f;
  for (int d = 0; d < 128; ++d)
    s += (j < 8 ? We[j * 128 + d] : be[d]) * Wm[d * 128 + c];
  wem9[idx] = s;
}

// out = LN((base? base : 0) + upd) * g + beta     (block = one 128-wide row)
__global__ void ln_k(const float* __restrict__ base, const float* __restrict__ upd,
                     const float* __restrict__ g, const float* __restrict__ beta,
                     float* __restrict__ out) {
  __shared__ float sm[128];
  int nrow = blockIdx.x, d = threadIdx.x;
  float x = upd[(size_t)nrow * 128 + d] + (base ? base[(size_t)nrow * 128 + d] : 0.f);
  sm[d] = x; __syncthreads();
  for (int s = 64; s > 0; s >>= 1) { if (d < s) sm[d] += sm[d + s]; __syncthreads(); }
  float mean = sm[0] * (1.f / 128.f);
  __syncthreads();
  float dx = x - mean;
  sm[d] = dx * dx; __syncthreads();
  for (int s = 64; s > 0; s >>= 1) { if (d < s) sm[d] += sm[d + s]; __syncthreads(); }
  float var = sm[0] * (1.f / 128.f);
  out[(size_t)nrow * 128 + d] = dx * rsqrtf(var + 1e-5f) * g[d] + beta[d];
}

// meta: sc_p = hpT_p[n].score ; a = softmax_p ; mix[n] = sum_p a_p * h_p[n]
__global__ void meta_mix_k(const float* __restrict__ hp0, const float* __restrict__ hp1,
                           const float* __restrict__ hp2, const float* __restrict__ score,
                           const float* __restrict__ h0, const float* __restrict__ h1,
                           const float* __restrict__ h2, float* __restrict__ mix) {
  __shared__ float sm[128];
  __shared__ float aP[3];
  int nrow = blockIdx.x, d = threadIdx.x;
  const float* hp[3] = {hp0, hp1, hp2};
  for (int p = 0; p < 3; ++p) {
    sm[d] = hp[p][(size_t)nrow * 128 + d] * score[d];
    __syncthreads();
    for (int s = 64; s > 0; s >>= 1) { if (d < s) sm[d] += sm[d + s]; __syncthreads(); }
    if (d == 0) aP[p] = sm[0];
    __syncthreads();
  }
  if (d == 0) {
    float m = fmaxf(aP[0], fmaxf(aP[1], aP[2]));
    float e0 = __expf(aP[0] - m), e1 = __expf(aP[1] - m), e2 = __expf(aP[2] - m);
    float inv = 1.f / (e0 + e1 + e2);
    aP[0] = e0 * inv; aP[1] = e1 * inv; aP[2] = e2 * inv;
  }
  __syncthreads();
  size_t o = (size_t)nrow * 128 + d;
  mix[o] = aP[0] * h0[o] + aP[1] * h1[o] + aP[2] * h2[o];
}

// ===========================================================================
extern "C" void kernel_launch(void* const* d_in, const int* in_sizes, int n_in,
                              void* d_out, int out_size, void* d_ws, size_t ws_size,
                              hipStream_t stream) {
  (void)in_sizes; (void)n_in; (void)out_size; (void)ws_size;
  auto F = [&](int i) { return (const float*)d_in[i]; };
  auto I = [&](int i) { return (const int*)d_in[i]; };

  // ---- workspace layout (floats); total ~7.22M floats (~29 MB) -----------
  float* ws = (float*)d_ws;
  const size_t SZ = (size_t)NS_ * D_;           // 512000
  float* t1   = ws;
  float* nsf  = ws + 1 * SZ;
  float* ndf  = ws + 2 * SZ;
  float* nmsg = ws + 3 * SZ;
  float* agg  = ws + 4 * SZ;
  float* upd  = ws + 5 * SZ;
  float* h_ss = ws + 6 * SZ;
  float* h_sb = ws + 7 * SZ;
  float* h_si = ws + 8 * SZ;
  float* hp0  = ws + 9 * SZ;
  float* hp1  = ws + 10 * SZ;
  float* hp2  = ws + 11 * SZ;
  float* mix  = ws + 12 * SZ;
  float* bx   = ws + 13 * SZ;                   // 256*128
  float* ix   = bx + (size_t)NB_ * D_;          // 128*128
  float* a_src = ix + (size_t)NI_ * D_;         // 4000
  float* a_dst = a_src + NS_;
  float* zseg  = a_dst + NS_;
  float* wem9  = zseg + NS_;                    // 9*128
  float* exb   = wem9 + 9 * 128;                // 500000 (max E)
  float* sx    = (float*)d_out;                 // final output buffer IS sx

  auto fill = [&](float* p, int n) {
    fill_zero_k<<<(n + 255) / 256, 256, 0, stream>>>(p, n);
  };

  // one relation forward (see top-of-file for leaf ordering)
  auto rel = [&](int base, bool has_edge, const float* src_x, int Nsrc,
                 const float* dst_x, int Ndst, const int* ei, int E,
                 const float* ea, float* out_buf) {
    const float* attnW = F(base + 0);
    const float* attnb = F(base + 1);
    const float* beta  = F(base + 2);
    const float* dstW  = F(base + 3);
    const float* dstb  = F(base + 4);
    const int og = has_edge ? 7 : 5;
    const float* g    = F(base + og);
    const float* msgW = F(base + og + 1);
    const float* msgb = F(base + og + 2);
    const float* outW = F(base + og + 3);
    const float* outb = F(base + og + 4);
    const float* srcW = F(base + og + 5);
    const float* srcb = F(base + og + 6);

    launch_gemm<128, false>(src_x, nullptr, srcW, srcb, nsf, Nsrc, 0, stream);
    launch_gemm<128, false>(dst_x, nullptr, dstW, dstb, ndf, Ndst, 0, stream);
    launch_gemm<128, false>(nsf,   nullptr, msgW, msgb, nmsg, Nsrc, 0, stream);
    dot_tanh_k<<<Nsrc, 128, 0, stream>>>(nsf, attnW,       a_src, Nsrc);
    dot_tanh_k<<<Ndst, 128, 0, stream>>>(ndf, attnW + 128, a_dst, Ndst);
    fill(zseg, Ndst);
    fill(agg, Ndst * 128);
    if (has_edge) {
      const float* edgeW = F(base + 5);
      const float* edgeb = F(base + 6);
      wem9_k<<<(9 * 128 + 255) / 256, 256, 0, stream>>>(edgeW, edgeb, msgW, wem9);
      edge_scoreA_ss_k<<<(E + 7) / 8, 256, 0, stream>>>(
          ei, E, ea, edgeW, edgeb, attnW + 256, attnb, a_src, a_dst, exb, zseg);
      edge_scatterB_ss_k<<<(E + 7) / 8, 256, 0, stream>>>(
          ei, E, ea, wem9, exb, zseg, nmsg, agg);
    } else {
      edge_scoreA_k<<<(E + 255) / 256, 256, 0, stream>>>(
          ei, E, a_src, a_dst, attnb, exb, zseg);
      edge_scatterB_k<<<(E * 32 + 255) / 256, 256, 0, stream>>>(
          ei, E, exb, zseg, nmsg, agg);
    }
    launch_gemm<256, true>(dst_x, agg, outW, outb, upd, Ndst, 0, stream);
    ln_k<<<Ndst, 128, 0, stream>>>(dst_x, upd, g, beta, out_buf);
  };

  // ---- encoders ----------------------------------------------------------
  launch_gemm<64,  false>(F(0), nullptr, F(167), F(168), t1, NS_, 1, stream);
  launch_gemm<128, false>(t1,   nullptr, F(169), F(170), sx, NS_, 0, stream);
  launch_gemm<32,  false>(F(1), nullptr, F(4),   F(5),   t1, NB_, 1, stream);
  launch_gemm<128, false>(t1,   nullptr, F(6),   F(7),   bx, NB_, 0, stream);
  launch_gemm<16,  false>(F(2), nullptr, F(8),   F(9),   t1, NI_, 1, stream);
  launch_gemm<128, false>(t1,   nullptr, F(10),  F(11),  ix, NI_, 0, stream);

  // ---- layers ------------------------------------------------------------
  const int L0 = 12, LSZ = 74;  // per-layer leaves: BS,II,IS,SB,SI (12 each) + SS (14)
  for (int li = 0; li < 2; ++li) {
    int lb  = L0 + li * LSZ;
    int bBS = lb + 0, bII = lb + 12, bIS = lb + 24, bSB = lb + 36, bSI = lb + 48, bSS = lb + 60;

    rel(bSS, true,  sx, NS_, sx, NS_, I(171), 500000, F(3),   h_ss);
    rel(bSB, false, bx, NB_, sx, NS_, I(172), 100000, nullptr, h_sb);
    rel(bSI, false, ix, NI_, sx, NS_, I(173), 100000, nullptr, h_si);

    // meta-path attention over {h_ss, h_sb, h_si}
    launch_gemm<128, false>(h_ss, nullptr, F(164), F(165), hp0, NS_, 2, stream);
    launch_gemm<128, false>(h_sb, nullptr, F(164), F(165), hp1, NS_, 2, stream);
    launch_gemm<128, false>(h_si, nullptr, F(164), F(165), hp2, NS_, 2, stream);
    meta_mix_k<<<NS_, 128, 0, stream>>>(hp0, hp1, hp2, F(166), h_ss, h_sb, h_si, mix);
    launch_gemm<128, false>(mix, nullptr, F(162), F(163), upd, NS_, 0, stream);
    ln_k<<<NS_, 128, 0, stream>>>(nullptr, upd, F(161), F(160), sx);  // no residual

    rel(bBS, false, sx, NS_, bx, NB_, I(174), 100000, nullptr, bx);
    rel(bIS, false, sx, NS_, ix, NI_, I(175), 100000, nullptr, ix);
    rel(bII, false, ix, NI_, ix, NI_, I(176), 20000,  nullptr, ix);
  }
  // sx (== d_out) now holds the final stock embeddings.
}
